// Model_52561809768908
// MI455X (gfx1250) — matrix operand order repaired and checked by reading, not compiled, not run on hardware
//
#include <hip/hip_runtime.h>
#include <hip/hip_bf16.h>
#include <math.h>

// ---- model dims ----
#define B_   32
#define T_   336
#define N_   321
#define L_   2
#define E_   4
#define F_   2048
#define P_   96
#define TP_  352     // T padded to multiple of 32 for bf16 WMMA K-loop
#define MROWS 32     // N rows per expert block
#define FC_  128     // F chunk width

typedef __bf16 bf16_t;
typedef bf16_t v16bf __attribute__((ext_vector_type(16)));
typedef bf16_t v8bf  __attribute__((ext_vector_type(8)));
typedef float  v8f   __attribute__((ext_vector_type(8)));

// Pack a v16bf fragment from two contiguous 16-byte loads (LDS or global).
__device__ inline v16bf pack16(const bf16_t* lo, const bf16_t* hi) {
    v8bf a = *(const v8bf*)lo;
    v8bf b = *(const v8bf*)hi;
    v16bf r;
#pragma unroll
    for (int i = 0; i < 8; ++i) { r[i] = a[i]; r[i + 8] = b[i]; }
    return r;
}

__device__ inline v8f wmma_bf16(v16bf a, v16bf b, v8f c) {
    return __builtin_amdgcn_wmma_f32_16x16x32_bf16(
        false, a, false, b, (short)0, c, false, false);
}

// ---------------------------------------------------------------------------
// Weight pre-passes: f32 -> bf16 with transpose into WMMA-B-friendly layouts.
// ---------------------------------------------------------------------------
// We1 (L,E,T,F) -> W1T (L,E,F,TP_) , zero-padded for t >= T
__global__ void convert_w1t(const float* __restrict__ src, bf16_t* __restrict__ dst) {
    size_t i = (size_t)blockIdx.x * blockDim.x + threadIdx.x;
    const size_t total = (size_t)L_ * E_ * F_ * TP_;
    if (i >= total) return;
    int t = (int)(i % TP_);
    size_t rest = i / TP_;
    int f = (int)(rest % F_);
    size_t le = rest / F_;
    float v = 0.0f;
    if (t < T_) v = src[(le * T_ + t) * F_ + f];
    dst[i] = (bf16_t)v;
}

// We2 (L,E,F,T) -> W2T (L,E,T,F)
__global__ void convert_w2t(const float* __restrict__ src, bf16_t* __restrict__ dst) {
    size_t i = (size_t)blockIdx.x * blockDim.x + threadIdx.x;
    const size_t total = (size_t)L_ * E_ * T_ * F_;
    if (i >= total) return;
    int f = (int)(i % F_);
    size_t rest = i / F_;
    int t = (int)(rest % T_);
    size_t le = rest / T_;
    dst[i] = (bf16_t)src[(le * F_ + f) * T_ + t];
}

// ---------------------------------------------------------------------------
// RevIN: S[b,n,t] = (x[b,t,n,0] - mu(b,n)) / sqrt(var(b,n) + 1e-5)
// ---------------------------------------------------------------------------
__global__ __launch_bounds__(128)
void revin_kernel(const float* __restrict__ x, float* __restrict__ S) {
    const int n = blockIdx.x, b = blockIdx.y, tid = threadIdx.x;
    __shared__ float xb[T_];
    __shared__ float red[128], red2[128];
    float s = 0.f, s2 = 0.f;
    for (int t = tid; t < T_; t += 128) {
        float v = x[(((size_t)b * T_ + t) * N_ + n) * 3 + 0];
        xb[t] = v; s += v; s2 += v * v;
    }
    red[tid] = s; red2[tid] = s2;
    __syncthreads();
    for (int off = 64; off > 0; off >>= 1) {
        if (tid < off) { red[tid] += red[tid + off]; red2[tid] += red2[tid + off]; }
        __syncthreads();
    }
    const float mu  = red[0] / T_;
    const float var = red2[0] / T_ - mu * mu;
    const float inv = 1.0f / sqrtf(var + 1e-5f);
    for (int t = tid; t < T_; t += 128)
        S[((size_t)b * N_ + n) * T_ + t] = (xb[t] - mu) * inv;
}

// ---------------------------------------------------------------------------
// m[b,t] = mean_n S[b,n,t]
// ---------------------------------------------------------------------------
__global__ void mean_kernel(const float* __restrict__ S, float* __restrict__ m) {
    const int b = blockIdx.x, t = threadIdx.x;
    float s = 0.f;
    for (int n = 0; n < N_; ++n) s += S[((size_t)b * N_ + n) * T_ + t];
    m[b * T_ + t] = s * (1.0f / N_);
}

// ---------------------------------------------------------------------------
// Top-2 gating + cv^2 balance loss. One wave32: lane b handles batch b.
// ---------------------------------------------------------------------------
__global__ void gate_kernel(const float* __restrict__ m, const float* __restrict__ Wg,
                            float* __restrict__ gates, float* __restrict__ loss_out,
                            int first) {
    __shared__ float imp[E_];
    const int b = threadIdx.x;
    if (b < E_) imp[b] = 0.0f;
    __syncthreads();
    float lg[E_];
#pragma unroll
    for (int e = 0; e < E_; ++e) lg[e] = 0.0f;
    for (int t = 0; t < T_; ++t) {
        float mv = m[b * T_ + t];
#pragma unroll
        for (int e = 0; e < E_; ++e) lg[e] += mv * Wg[t * E_ + e];
    }
    int i1 = 0; float v1 = lg[0];
#pragma unroll
    for (int e = 1; e < E_; ++e) if (lg[e] > v1) { v1 = lg[e]; i1 = e; }
    int i2 = 0; float v2 = -1e30f;
#pragma unroll
    for (int e = 0; e < E_; ++e) if (e != i1 && lg[e] > v2) { v2 = lg[e]; i2 = e; }
    const float e2 = expf(v2 - v1);
    const float g1 = 1.0f / (1.0f + e2);
    const float g2 = e2 / (1.0f + e2);
#pragma unroll
    for (int e = 0; e < E_; ++e) gates[b * E_ + e] = 0.0f;
    gates[b * E_ + i1] = g1;
    gates[b * E_ + i2] = g2;
    atomicAdd(&imp[i1], g1);
    atomicAdd(&imp[i2], g2);
    __syncthreads();
    if (b == 0) {
        float s = 0.f;
        for (int e = 0; e < E_; ++e) s += imp[e];
        const float mean = s / E_;
        float var = 0.f;
        for (int e = 0; e < E_; ++e) { float d = imp[e] - mean; var += d * d; }
        var /= E_;
        const float contrib = 0.01f * (var / (mean * mean + 1e-10f));
        if (first) *loss_out = contrib; else *loss_out += contrib;
    }
}

// ---------------------------------------------------------------------------
// Fused MoE expert layer: S[b, band, :] += sum_e gate[b,e] *
//      ( relu(X @ We1[e] + be1[e]) @ We2[e] + be2[e] )
// Block = (n-band of 32, b); 16 waves; bf16 WMMA for both GEMMs.
// B fragments stream from pre-transposed global weights (L2-resident);
// A/B fragments are software-pipelined one K-step ahead of each WMMA.
// ---------------------------------------------------------------------------
__global__ __launch_bounds__(512)
void expert_kernel(const bf16_t* __restrict__ W1T,  // (E,F,TP_) bf16, this layer
                   const bf16_t* __restrict__ W2T,  // (E,T,F)  bf16, this layer
                   const float*  __restrict__ be1,  // (E,F)
                   const float*  __restrict__ be2,  // (E,T)
                   const float*  __restrict__ gates,// (B,E)
                   float* __restrict__ S)           // (B,N,T) in/out
{
    __shared__ __align__(16) bf16_t Xs[MROWS][360];   // X band, K-padded, bf16
    __shared__ __align__(16) bf16_t Hs[MROWS][136];   // H chunk (32 x 128)

    const int b    = blockIdx.y;
    const int n0   = blockIdx.x * MROWS;
    const int tid  = threadIdx.x;
    const int lane = tid & 31;
    const int w    = tid >> 5;                 // wave 0..15
    const int half = (lane >= 16) ? 1 : 0;
    const int ln   = lane & 15;
    const int mt   = w & 1;                    // phase-1 tile assignment
    const int ft   = w >> 1;

    // stage X band (f32 state -> bf16 LDS), zero-pad rows >= N and t >= T
    for (int idx = tid; idx < MROWS * TP_; idx += 512) {
        int r = idx / TP_, t = idx % TP_;
        int n = n0 + r;
        float v = 0.0f;
        if (n < N_ && t < T_) v = S[((size_t)b * N_ + n) * T_ + t];
        Xs[r][t] = (bf16_t)v;
    }
    __syncthreads();

    // phase-2 tile ownership (wave-uniform): q = w, w+16, w+32 (<42)
    const int ntiles = (w < 10) ? 3 : 2;
    const int nsteps = ntiles * 4;

    v8f   acc[3];
    float fin[3][8];
#pragma unroll
    for (int j = 0; j < 3; ++j)
#pragma unroll
        for (int r = 0; r < 8; ++r) fin[j][r] = 0.0f;

    for (int e = 0; e < E_; ++e) {
        const float gate = gates[b * E_ + e];
        if (gate == 0.0f) continue;            // block-uniform: skip unrouted experts
        const bf16_t* W1e = W1T + (size_t)e * F_ * TP_;
        const bf16_t* W2e = W2T + (size_t)e * T_ * F_;

#pragma unroll
        for (int j = 0; j < 3; ++j)
#pragma unroll
            for (int r = 0; r < 8; ++r) acc[j][r] = 0.0f;

        for (int f0 = 0; f0 < F_; f0 += FC_) {
            // ---- phase 1: H(32 x 128) = relu(X @ We1[:, f0:f0+128] + be1)
            // One-deep software pipeline: fragments for step k+1 issued before
            // the WMMA consuming step k, so waits are on iteration-old loads.
            const bf16_t* b1 = W1e + (size_t)(f0 + ft * 16 + ln) * TP_ + half * 8;
            const bf16_t* a1 = &Xs[mt * 16 + ln][half * 8];
            v8f c1;
#pragma unroll
            for (int r = 0; r < 8; ++r) c1[r] = 0.0f;

            v16bf aC = pack16(a1, a1 + 16);
            v16bf bC = pack16(b1, b1 + 16);
#pragma unroll
            for (int ks = 0; ks < 11; ++ks) {
                v16bf aN, bN;
                if (ks < 10) {
                    const int k1 = (ks + 1) * 32;
                    aN = pack16(a1 + k1, a1 + k1 + 16);
                    bN = pack16(b1 + k1, b1 + k1 + 16);
                }
                c1 = wmma_bf16(aC, bC, c1);
                if (ks < 10) { aC = aN; bC = bN; }
            }
            __syncthreads();                   // prior phase-2 readers of Hs done
            {
                const float bias = be1[e * F_ + f0 + ft * 16 + ln];
#pragma unroll
                for (int r = 0; r < 8; ++r) {
                    float hv = c1[r] + bias;
                    hv = hv > 0.f ? hv : 0.f;
                    Hs[mt * 16 + r + half * 8][ft * 16 + ln] = (bf16_t)hv;
                }
            }
            __syncthreads();                   // Hs visible to all waves

            // ---- phase 2: Eo(32 x 336) += H @ We2[f0:f0+128, :]
            // Flattened (tile x k-step) pipeline, prefetch across tile bounds.
            const bf16_t* a2p[3];
            const bf16_t* b2p[3];
#pragma unroll
            for (int j = 0; j < 3; ++j) {
                const int q = w + 16 * j;
                if (q < 42) {
                    a2p[j] = &Hs[(q & 1) * 16 + ln][half * 8];
                    b2p[j] = W2e + (size_t)((q >> 1) * 16 + ln) * F_
                             + f0 + half * 8;
                } else {
                    a2p[j] = a2p[0];
                    b2p[j] = b2p[0];
                }
            }
            v16bf a2C = pack16(a2p[0], a2p[0] + 16);
            v16bf b2C = pack16(b2p[0], b2p[0] + 16);
#pragma unroll
            for (int s = 0; s < 12; ++s) {
                if (s >= nsteps) break;        // wave-uniform
                const int s1 = s + 1;
                v16bf aN, bN;
                if (s1 < nsteps) {
                    const int j1 = s1 >> 2;
                    const int k1 = (s1 & 3) * 32;
                    aN = pack16(a2p[j1] + k1, a2p[j1] + k1 + 16);
                    bN = pack16(b2p[j1] + k1, b2p[j1] + k1 + 16);
                }
                acc[s >> 2] = wmma_bf16(a2C, b2C, acc[s >> 2]);
                if (s1 < nsteps) { a2C = aN; b2C = bN; }
            }
        }
        // gate-weighted fold with be2 bias
#pragma unroll
        for (int j = 0; j < 3; ++j) {
            const int q = w + 16 * j;
            if (q >= 42) break;
            const int tt = q >> 1;
            const float b2v = be2[e * T_ + tt * 16 + ln];
#pragma unroll
            for (int r = 0; r < 8; ++r) fin[j][r] += gate * (acc[j][r] + b2v);
        }
    }

    // residual write-back (only this block touches this band)
#pragma unroll
    for (int j = 0; j < 3; ++j) {
        const int q = w + 16 * j;
        if (q >= 42) continue;
        const int tt  = q >> 1;
        const int mt2 = q & 1;
        const int t = tt * 16 + ln;
#pragma unroll
        for (int r = 0; r < 8; ++r) {
            const int n = n0 + mt2 * 16 + r + half * 8;
            if (n < N_) {
                const size_t idx = ((size_t)b * N_ + n) * T_ + t;
                S[idx] = S[idx] + fin[j][r];
            }
        }
    }
}

// ---------------------------------------------------------------------------
// Projection head: tanh(S @ P1w + P1b) @ P2w + P2b -> mean / softplus std
// ---------------------------------------------------------------------------
__global__ __launch_bounds__(256)
void head_kernel(const float* __restrict__ S,
                 const float* __restrict__ P1w, const float* __restrict__ P1b,
                 const float* __restrict__ P2w, const float* __restrict__ P2b,
                 float* __restrict__ out) {
    __shared__ float Ob[16][T_];
    __shared__ float Hp[16][P_];
    const int n0 = blockIdx.x * 16, b = blockIdx.y, tid = threadIdx.x;

    for (int idx = tid; idx < 16 * T_; idx += 256) {
        int r = idx / T_, t = idx % T_;
        int n = n0 + r;
        Ob[r][t] = (n < N_) ? S[((size_t)b * N_ + n) * T_ + t] : 0.0f;
    }
    __syncthreads();
    for (int idx = tid; idx < 16 * P_; idx += 256) {
        int r = idx / P_, p = idx % P_;
        float s = P1b[p];
        for (int t = 0; t < T_; ++t) s += Ob[r][t] * P1w[t * P_ + p];
        Hp[r][p] = tanhf(s);
    }
    __syncthreads();
    const size_t stdoff = (size_t)B_ * P_ * N_ + 1;
    for (int idx = tid; idx < 16 * 2 * P_; idx += 256) {
        int r = idx / (2 * P_), c = idx % (2 * P_);
        int n = n0 + r;
        if (n >= N_) continue;
        float s = P2b[c];
        for (int k = 0; k < P_; ++k) s += Hp[r][k] * P2w[k * (2 * P_) + c];
        int p = c >> 1;
        size_t o = ((size_t)b * P_ + p) * N_ + n;
        if ((c & 1) == 0) out[o] = s;
        else out[stdoff + o] = ((s > 20.0f) ? s : log1pf(expf(s))) + 1e-6f;
    }
}

// ---------------------------------------------------------------------------
extern "C" void kernel_launch(void* const* d_in, const int* in_sizes, int n_in,
                              void* d_out, int out_size, void* d_ws, size_t ws_size,
                              hipStream_t stream) {
    const float* x   = (const float*)d_in[0];
    // d_in[1..10] (embedding + attention weights) are dead code in the reference.
    const float* Wg  = (const float*)d_in[11];
    const float* We1 = (const float*)d_in[12];
    const float* be1 = (const float*)d_in[13];
    const float* We2 = (const float*)d_in[14];
    const float* be2 = (const float*)d_in[15];
    const float* P1w = (const float*)d_in[16];
    const float* P1b = (const float*)d_in[17];
    const float* P2w = (const float*)d_in[18];
    const float* P2b = (const float*)d_in[19];
    float* out = (float*)d_out;

    char* ws = (char*)d_ws;
    size_t off = 0;
    auto carve = [&](size_t bytes) -> char* {
        char* p = ws + off;
        off += (bytes + 255) & ~(size_t)255;
        return p;
    };
    float*  S     = (float*)carve(sizeof(float) * (size_t)B_ * N_ * T_);
    float*  m     = (float*)carve(sizeof(float) * (size_t)B_ * T_);
    float*  gates = (float*)carve(sizeof(float) * (size_t)B_ * E_);
    bf16_t* W1bfT = (bf16_t*)carve(sizeof(bf16_t) * (size_t)L_ * E_ * F_ * TP_);
    bf16_t* W2bfT = (bf16_t*)carve(sizeof(bf16_t) * (size_t)L_ * E_ * T_ * F_);

    {
        const size_t n1 = (size_t)L_ * E_ * F_ * TP_;
        convert_w1t<<<(unsigned)((n1 + 255) / 256), 256, 0, stream>>>(We1, W1bfT);
        const size_t n2 = (size_t)L_ * E_ * T_ * F_;
        convert_w2t<<<(unsigned)((n2 + 255) / 256), 256, 0, stream>>>(We2, W2bfT);
    }

    revin_kernel<<<dim3(N_, B_), 128, 0, stream>>>(x, S);

    for (int l = 0; l < L_; ++l) {
        mean_kernel<<<B_, T_, 0, stream>>>(S, m);
        gate_kernel<<<1, 32, 0, stream>>>(m, Wg + (size_t)l * T_ * E_, gates,
                                          out + (size_t)B_ * P_ * N_, l == 0 ? 1 : 0);
        expert_kernel<<<dim3(11, B_), 512, 0, stream>>>(
            W1bfT + (size_t)l * E_ * F_ * TP_, W2bfT + (size_t)l * E_ * T_ * F_,
            be1 + (size_t)l * E_ * F_, be2 + (size_t)l * E_ * T_,
            gates, S);
    }

    head_kernel<<<dim3(21, B_), 256, 0, stream>>>(S, P1w, P1b, P2w, P2b, out);
}
